// ML_VAE_45165876084803
// MI455X (gfx1250) — matrix operand order repaired and checked by reading, not compiled, not run on hardware
//
#include <hip/hip_runtime.h>

// ---------------------------------------------------------------------------
// MI455X (gfx1250) VAE forward. All heavy GEMM-shaped math on
// v_wmma_f32_16x16x32_bf16 (wave32, 16x16 tiles), f32 accumulate.
// Head GEMMs: split-K + 32x64 register tiles (occupancy + intensity).
// Convs 32->64 / 64->32: tap-decomposed implicit GEMM, one wave per 32-wide
// spatial row so weight fragments are reused across 2 A-tiles.
// ---------------------------------------------------------------------------

typedef __attribute__((ext_vector_type(16))) __bf16 v16bf;
typedef __attribute__((ext_vector_type(8)))  __bf16 v8bf;
typedef __attribute__((ext_vector_type(8)))  float  v8f;

#define N_  512
#define S_  32
#define P_  (N_ * S_ * S_)   // 524288 spatial positions
#define L_  1024
#define F_  32768
#define H_  512

__device__ __forceinline__ float lrelu_(float v) { return v >= 0.f ? v : 0.01f * v; }

__device__ __forceinline__ v16bf frag_zero() {
    v16bf z;
#pragma unroll
    for (int i = 0; i < 16; ++i) z[i] = (__bf16)0.0f;
    return z;
}

// A fragment (16-bit A 16x32 layout): lane half h holds K = h*8..h*8+7 (v0-3)
// and K = 16+h*8..16+h*8+7 (v4-7), row = lane&15.
__device__ __forceinline__ v16bf load_fragA(const __bf16* __restrict__ row, int half) {
    v8bf lo = *(const v8bf*)(row + half * 8);
    v8bf hi = *(const v8bf*)(row + 16 + half * 8);
    return __builtin_shufflevector(lo, hi, 0,1,2,3,4,5,6,7,8,9,10,11,12,13,14,15);
}

// B fragment: col = lane&15, same K order as A: half h -> K=h*8..h*8+7, 16+h*8..16+h*8+7.
__device__ __forceinline__ v16bf load_fragB(const __bf16* __restrict__ row, int half) {
    v8bf lo = *(const v8bf*)(row + half * 8);
    v8bf hi = *(const v8bf*)(row + 16 + half * 8);
    return __builtin_shufflevector(lo, hi, 0,1,2,3,4,5,6,7,8,9,10,11,12,13,14,15);
}

#define WMMA_BF16(a, b, c) \
    __builtin_amdgcn_wmma_f32_16x16x32_bf16(false, (a), false, (b), (short)0, (c), false, false)

// ---------------------------------------------------------------------------
// Split-K GEMM: part[z, m, n] = sum_{k in slice z} A[m,k] * W[n,k]
// A: [M,K] bf16 row-major.  W: [Nout,K] bf16 row-major.
// 1 wave = 32x64 tile (2 A frags x 4 B frags = 8 WMMA per K-step).
// K = gridDim.z * Kslice.
// ---------------------------------------------------------------------------
__global__ __launch_bounds__(32)
void k_gemm_splitk(const __bf16* __restrict__ A, const __bf16* __restrict__ W,
                   float* __restrict__ part, int M, int Nout, int Kslice) {
    const int m0 = blockIdx.x * 32;
    const int n0 = blockIdx.y * 64;
    const int K  = (int)gridDim.z * Kslice;
    const int kbase = blockIdx.z * Kslice;
    const int lane = threadIdx.x;
    const int r = lane & 15, half = lane >> 4;

    const __bf16* Ar0 = A + (size_t)(m0 + r) * K + kbase;
    const __bf16* Ar1 = A + (size_t)(m0 + 16 + r) * K + kbase;
    const __bf16* Wr0 = W + (size_t)(n0 +  0 + r) * K + kbase;
    const __bf16* Wr1 = W + (size_t)(n0 + 16 + r) * K + kbase;
    const __bf16* Wr2 = W + (size_t)(n0 + 32 + r) * K + kbase;
    const __bf16* Wr3 = W + (size_t)(n0 + 48 + r) * K + kbase;

    v8f acc0[4], acc1[4];
#pragma unroll
    for (int nb = 0; nb < 4; ++nb)
#pragma unroll
        for (int j = 0; j < 8; ++j) { acc0[nb][j] = 0.f; acc1[nb][j] = 0.f; }

    for (int k0 = 0; k0 < Kslice; k0 += 32) {
        v16bf a0 = load_fragA(Ar0 + k0, half);
        v16bf a1 = load_fragA(Ar1 + k0, half);
        v16bf b0 = load_fragB(Wr0 + k0, half);
        v16bf b1 = load_fragB(Wr1 + k0, half);
        v16bf b2 = load_fragB(Wr2 + k0, half);
        v16bf b3 = load_fragB(Wr3 + k0, half);
        acc0[0] = WMMA_BF16(a0, b0, acc0[0]);
        acc1[0] = WMMA_BF16(a1, b0, acc1[0]);
        acc0[1] = WMMA_BF16(a0, b1, acc0[1]);
        acc1[1] = WMMA_BF16(a1, b1, acc1[1]);
        acc0[2] = WMMA_BF16(a0, b2, acc0[2]);
        acc1[2] = WMMA_BF16(a1, b2, acc1[2]);
        acc0[3] = WMMA_BF16(a0, b3, acc0[3]);
        acc1[3] = WMMA_BF16(a1, b3, acc1[3]);
    }

    float* pbase = part + (size_t)blockIdx.z * M * Nout;
#pragma unroll
    for (int nb = 0; nb < 4; ++nb) {
        int col = n0 + nb * 16 + r;
#pragma unroll
        for (int j = 0; j < 8; ++j) {
            int m = m0 + half * 8 + j;
            pbase[(size_t)m * Nout + col] = acc0[nb][j];
            pbase[(size_t)(m + 16) * Nout + col] = acc1[nb][j];
        }
    }
}

// reduce split-K partials + bias + lrelu (+ optional bf16 convert)
template <bool OUT_BF16>
__global__ void k_gemm_fin(const float* __restrict__ part, const float* __restrict__ bias,
                           void* __restrict__ out, int M, int Nout, int nmask, int nsl) {
    int idx = blockIdx.x * blockDim.x + threadIdx.x;   // M*Nout total (exact)
    int col = idx & nmask;
    float s = bias[col];
    size_t stride = (size_t)M * Nout;
    for (int t = 0; t < nsl; ++t) s += part[(size_t)t * stride + idx];
    float v = lrelu_(s);
    if (OUT_BF16) ((__bf16*)out)[idx] = (__bf16)v;
    else          ((float*)out)[idx]  = v;
}

// ---------------------------------------------------------------------------
// conv 3x3 SAME, Cin=32 -> Cout=64, tap-decomposed implicit GEMM (K=32/tap).
// in: NHWC bf16 [P_,32]; wt: [tap][64][32] bf16; out: f32 [P_,64] (pre-BN).
// One wave = one full 32-wide spatial row x 64 out-ch (2 A-tiles share B frags).
// ---------------------------------------------------------------------------
__global__ __launch_bounds__(32)
void k_conv2_wmma(const __bf16* __restrict__ in, const __bf16* __restrict__ wt,
                  const float* __restrict__ bias, float* __restrict__ out) {
    const int p0 = blockIdx.x * 32;                 // row-aligned
    const int n = p0 >> 10, yy = (p0 & 1023) >> 5;
    const int lane = threadIdx.x, r = lane & 15, half = lane >> 4;

    v8f acc0[4], acc1[4];
#pragma unroll
    for (int nb = 0; nb < 4; ++nb)
#pragma unroll
        for (int j = 0; j < 8; ++j) { acc0[nb][j] = 0.f; acc1[nb][j] = 0.f; }

#pragma unroll
    for (int tap = 0; tap < 9; ++tap) {
        const int dy = tap / 3 - 1, dx = tap % 3 - 1;
        const int sy = yy + dy;
        const bool syok = (sy >= 0 && sy < 32);
        const __bf16* rowb = in + ((size_t)n * 1024 + (size_t)sy * 32) * 32;
        const int sx0 = r + dx, sx1 = 16 + r + dx;
        v16bf a0 = (syok && sx0 >= 0 && sx0 < 32) ? load_fragA(rowb + (size_t)sx0 * 32, half) : frag_zero();
        v16bf a1 = (syok && sx1 >= 0 && sx1 < 32) ? load_fragA(rowb + (size_t)sx1 * 32, half) : frag_zero();
        const __bf16* wtap = wt + (size_t)tap * 64 * 32;
#pragma unroll
        for (int nb = 0; nb < 4; ++nb) {
            v16bf b = load_fragB(wtap + (size_t)(nb * 16 + r) * 32, half);
            acc0[nb] = WMMA_BF16(a0, b, acc0[nb]);
            acc1[nb] = WMMA_BF16(a1, b, acc1[nb]);
        }
    }

#pragma unroll
    for (int nb = 0; nb < 4; ++nb) {
        int col = nb * 16 + r;
        float bv = bias[col];
#pragma unroll
        for (int j = 0; j < 8; ++j) {
            int m = half * 8 + j;
            out[(size_t)(p0 + m) * 64 + col]      = acc0[nb][j] + bv;
            out[(size_t)(p0 + 16 + m) * 64 + col] = acc1[nb][j] + bv;
        }
    }
}

// conv 3x3 SAME, Cin=64 -> Cout=32. K=64/tap -> 2 K-steps. wt: [tap][32][64].
__global__ __launch_bounds__(32)
void k_conv3_wmma(const __bf16* __restrict__ in, const __bf16* __restrict__ wt,
                  const float* __restrict__ bias, float* __restrict__ out) {
    const int p0 = blockIdx.x * 32;
    const int n = p0 >> 10, yy = (p0 & 1023) >> 5;
    const int lane = threadIdx.x, r = lane & 15, half = lane >> 4;

    v8f acc0[2], acc1[2];
#pragma unroll
    for (int nb = 0; nb < 2; ++nb)
#pragma unroll
        for (int j = 0; j < 8; ++j) { acc0[nb][j] = 0.f; acc1[nb][j] = 0.f; }

#pragma unroll
    for (int tap = 0; tap < 9; ++tap) {
        const int dy = tap / 3 - 1, dx = tap % 3 - 1;
        const int sy = yy + dy;
        const bool syok = (sy >= 0 && sy < 32);
        const __bf16* rowb = in + ((size_t)n * 1024 + (size_t)sy * 32) * 64;
        const int sx0 = r + dx, sx1 = 16 + r + dx;
        const bool ok0 = syok && sx0 >= 0 && sx0 < 32;
        const bool ok1 = syok && sx1 >= 0 && sx1 < 32;
#pragma unroll
        for (int kh = 0; kh < 2; ++kh) {
            v16bf a0 = ok0 ? load_fragA(rowb + (size_t)sx0 * 64 + kh * 32, half) : frag_zero();
            v16bf a1 = ok1 ? load_fragA(rowb + (size_t)sx1 * 64 + kh * 32, half) : frag_zero();
#pragma unroll
            for (int nb = 0; nb < 2; ++nb) {
                v16bf b = load_fragB(wt + (size_t)(tap * 32 + nb * 16 + r) * 64 + kh * 32, half);
                acc0[nb] = WMMA_BF16(a0, b, acc0[nb]);
                acc1[nb] = WMMA_BF16(a1, b, acc1[nb]);
            }
        }
    }

#pragma unroll
    for (int nb = 0; nb < 2; ++nb) {
        int col = nb * 16 + r;
        float bv = bias[col];
#pragma unroll
        for (int j = 0; j < 8; ++j) {
            int m = half * 8 + j;
            out[(size_t)(p0 + m) * 32 + col]      = acc0[nb][j] + bv;
            out[(size_t)(p0 + 16 + m) * 32 + col] = acc1[nb][j] + bv;
        }
    }
}

// ---------------------------------------------------------------------------
// Small direct convs + BN stats/apply + misc
// ---------------------------------------------------------------------------
__global__ void k_conv1(const float* __restrict__ x, const float* __restrict__ w /*[32,1,3,3]*/,
                        const float* __restrict__ b, float* __restrict__ y /*[P_,32]*/) {
    __shared__ float wl[32 * 9];
    __shared__ float bl[32];
    for (int t = threadIdx.x; t < 32 * 9; t += blockDim.x) wl[t] = w[t];
    if (threadIdx.x < 32) bl[threadIdx.x] = b[threadIdx.x];
    __syncthreads();
    int p = blockIdx.x * blockDim.x + threadIdx.x;
    int n = p >> 10, rem = p & 1023, yy = rem >> 5, xx = rem & 31;
    float t[9];
#pragma unroll
    for (int tap = 0; tap < 9; ++tap) {
        int dy = tap / 3 - 1, dx = tap % 3 - 1;
        int sy = yy + dy, sx = xx + dx;
        t[tap] = (sy >= 0 && sy < 32 && sx >= 0 && sx < 32)
                 ? x[(size_t)n * 1024 + sy * 32 + sx] : 0.f;
    }
#pragma unroll 4
    for (int o = 0; o < 32; ++o) {
        float acc = bl[o];
#pragma unroll
        for (int tap = 0; tap < 9; ++tap) acc += wl[o * 9 + tap] * t[tap];
        y[(size_t)p * 32 + o] = acc;
    }
}

__global__ void k_conv3_dec(const __bf16* __restrict__ in /*[P_,64]*/,
                            const float* __restrict__ w /*[1,64,3,3]*/,
                            const float* __restrict__ b, float* __restrict__ y /*[P_]*/) {
    __shared__ float wl[9 * 64];   // wl[tap*64+ic]
    for (int t = threadIdx.x; t < 576; t += blockDim.x) {
        int tap = t / 64, ic = t - tap * 64;
        wl[t] = w[ic * 9 + tap];
    }
    __syncthreads();
    int p = blockIdx.x * blockDim.x + threadIdx.x;
    int n = p >> 10, rem = p & 1023, yy = rem >> 5, xx = rem & 31;
    float acc = b[0];
#pragma unroll
    for (int tap = 0; tap < 9; ++tap) {
        int dy = tap / 3 - 1, dx = tap % 3 - 1;
        int sy = yy + dy, sx = xx + dx;
        if (sy >= 0 && sy < 32 && sx >= 0 && sx < 32) {
            const __bf16* row = in + ((size_t)n * 1024 + sy * 32 + sx) * 64;
#pragma unroll 8
            for (int ic = 0; ic < 64; ++ic) acc += (float)row[ic] * wl[tap * 64 + ic];
        }
    }
    y[p] = acc;
}

// per-channel sum/sumsq, fully coalesced: channel = tid mod C (C | 256, C | stride)
__global__ void k_stats(const float* __restrict__ y, int C, float* __restrict__ stats) {
    __shared__ float s0[256], s1[256];
    float a = 0.f, b = 0.f;
    const size_t total = (size_t)P_ * C;
    for (size_t e = (size_t)blockIdx.x * 256 + threadIdx.x; e < total;
         e += (size_t)gridDim.x * 256) {
        float v = y[e];
        a += v; b += v * v;
    }
    s0[threadIdx.x] = a; s1[threadIdx.x] = b;
    __syncthreads();
    for (int o = 128; o >= C; o >>= 1) {
        if (threadIdx.x < o) { s0[threadIdx.x] += s0[threadIdx.x + o]; s1[threadIdx.x] += s1[threadIdx.x + o]; }
        __syncthreads();
    }
    if (threadIdx.x < C) {
        atomicAdd(&stats[threadIdx.x * 2 + 0], s0[threadIdx.x]);
        atomicAdd(&stats[threadIdx.x * 2 + 1], s1[threadIdx.x]);
    }
}

template <bool OUT_BF16>
__global__ void k_bn_lrelu(const float* __restrict__ y, const float* __restrict__ stats,
                           int cmask, void* __restrict__ out, size_t total) {
    size_t idx = (size_t)blockIdx.x * blockDim.x + threadIdx.x;
    if (idx >= total) return;
    int c = (int)(idx & (size_t)cmask);
    float mean = stats[c * 2 + 0] * (1.f / (float)P_);
    float var  = stats[c * 2 + 1] * (1.f / (float)P_) - mean * mean;
    float v = (y[idx] - mean) * rsqrtf(var + 1e-5f);
    v = lrelu_(v);
    if (OUT_BF16) ((__bf16*)out)[idx] = (__bf16)v;
    else          ((float*)out)[idx] = v;
}

// weight converters ---------------------------------------------------------
__global__ void k_cvt(const float* __restrict__ src, __bf16* __restrict__ dst, int n) {
    int i = blockIdx.x * blockDim.x + threadIdx.x;
    if (i < n) dst[i] = (__bf16)src[i];
}
// OIHW [O,I,3,3] f32 -> [tap][O][I] bf16
__global__ void k_cvt_convw(const float* __restrict__ src, __bf16* __restrict__ dst, int O, int I) {
    int idx = blockIdx.x * blockDim.x + threadIdx.x;
    int OI = O * I;
    if (idx >= 9 * OI) return;
    int tap = idx / OI, rem = idx - tap * OI;
    int o = rem / I, i = rem - o * I;
    dst[idx] = (__bf16)src[(size_t)(o * I + i) * 9 + tap];
}
// W1 [H,F] (F = c*1024 + s) f32 -> bf16 [H][s*32+c] via LDS 32x32 tile transpose
__global__ void k_cvt_w1(const float* __restrict__ src, __bf16* __restrict__ dst) {
    __shared__ float t[32][33];
    const int bx = blockIdx.x;            // h*32 + s-tile
    const int h = bx >> 5, s0 = (bx & 31) << 5;
    const int tid = threadIdx.x;          // 256 threads, 4 elems each
#pragma unroll
    for (int k = 0; k < 4; ++k) {
        int e = tid + k * 256;            // 0..1023
        int c = e >> 5, ss = e & 31;
        t[c][ss] = src[((size_t)h << 15) + (size_t)c * 1024 + (s0 + ss)];
    }
    __syncthreads();
#pragma unroll
    for (int k = 0; k < 4; ++k) {
        int e = tid + k * 256;
        int ss = e >> 5, c = e & 31;
        dst[((size_t)h << 15) + (size_t)(s0 + ss) * 32 + c] = (__bf16)t[c][ss];
    }
}

// product-of-experts group posterior (block per latent dim l)
__global__ void k_poe(const float* __restrict__ mu_s, const float* __restrict__ lv_s,
                      float* __restrict__ gmu, float* __restrict__ glv) {
    int l = blockIdx.x;
    __shared__ float sp[256], sm[256];
    float p = 0.f, m = 0.f;
    for (int n = threadIdx.x; n < N_; n += 256) {
        float prec = expf(-lv_s[(size_t)n * L_ + l]);
        p += prec;
        m += mu_s[(size_t)n * L_ + l] * prec;
    }
    sp[threadIdx.x] = p; sm[threadIdx.x] = m;
    __syncthreads();
    for (int o = 128; o > 0; o >>= 1) {
        if (threadIdx.x < o) { sp[threadIdx.x] += sp[threadIdx.x + o]; sm[threadIdx.x] += sm[threadIdx.x + o]; }
        __syncthreads();
    }
    if (threadIdx.x == 0) {
        float gv = 1.f / sp[0];
        gmu[l] = sm[0] * gv;
        glv[l] = logf(gv);
    }
}

__global__ void k_reparam(const float* __restrict__ eps_c, const float* __restrict__ eps_s,
                          const float* __restrict__ mu_c, const float* __restrict__ lv_c,
                          const float* __restrict__ gmu, const float* __restrict__ glv,
                          float* __restrict__ zc, float* __restrict__ zs) {
    size_t idx = (size_t)blockIdx.x * blockDim.x + threadIdx.x;
    int l = (int)(idx & (L_ - 1));
    zc[idx] = eps_c[idx] * expf(0.5f * lv_c[idx]) + mu_c[idx];
    zs[idx] = eps_s[idx] * expf(0.5f * glv[l]) + gmu[l];
}

__global__ void k_kl1(const float* __restrict__ mu, const float* __restrict__ lv,
                      float* __restrict__ scal) {
    __shared__ float s[256];
    float a = 0.f;
    for (size_t i = (size_t)blockIdx.x * blockDim.x + threadIdx.x; i < (size_t)N_ * L_;
         i += (size_t)gridDim.x * blockDim.x) {
        float m = mu[i], l = lv[i], e = expf(l);
        a += -0.5f * (1.f + l) + 0.5f * (m * m + e * e);
    }
    s[threadIdx.x] = a;
    __syncthreads();
    for (int o = 128; o > 0; o >>= 1) {
        if (threadIdx.x < o) s[threadIdx.x] += s[threadIdx.x + o];
        __syncthreads();
    }
    if (threadIdx.x == 0) atomicAdd(scal, s[0]);
}

__global__ void k_kl2(const float* __restrict__ gmu, const float* __restrict__ glv,
                      float* __restrict__ scal) {
    __shared__ float s[256];
    float a = 0.f;
    for (int l = threadIdx.x; l < L_; l += 256) {
        float m = gmu[l], v = glv[l], e = expf(v);
        a += -0.5f * (1.f + v) + 0.5f * (m * m + e * e);
    }
    s[threadIdx.x] = a;
    __syncthreads();
    for (int o = 128; o > 0; o >>= 1) {
        if (threadIdx.x < o) s[threadIdx.x] += s[threadIdx.x + o];
        __syncthreads();
    }
    if (threadIdx.x == 0) scal[1] = s[0];
}

__global__ void k_final(const float* __restrict__ x, const float* __restrict__ outc,
                        const float* __restrict__ outs, const float* __restrict__ scal,
                        float* __restrict__ dout) {
    size_t idx = (size_t)blockIdx.x * blockDim.x + threadIdx.x;
    float kl1 = scal[0], kl2 = scal[1];
    float o = outc[idx] * fmaxf(outs[idx], 0.f);
    float d = expf(x[idx]) - expf(o);
    dout[idx] = -kl1 + kl2 + d * d;
    dout[(size_t)P_ + idx] = o;
}

__global__ void k_zero(float* __restrict__ p, int n) {
    int i = blockIdx.x * blockDim.x + threadIdx.x;
    if (i < n) p[i] = 0.f;
}

// ---------------------------------------------------------------------------
extern "C" void kernel_launch(void* const* d_in, const int* in_sizes, int n_in,
                              void* d_out, int out_size, void* d_ws, size_t ws_size,
                              hipStream_t stream) {
    (void)in_sizes; (void)n_in; (void)out_size; (void)ws_size;
    const float* x        = (const float*)d_in[0];
    const float* eps_c    = (const float*)d_in[1];
    const float* eps_s    = (const float*)d_in[2];
    const float* enc_cw1  = (const float*)d_in[3];
    const float* enc_cb1  = (const float*)d_in[4];
    const float* enc_cw2  = (const float*)d_in[5];
    const float* enc_cb2  = (const float*)d_in[6];
    const float* enc_cw3  = (const float*)d_in[7];
    const float* enc_cb3  = (const float*)d_in[8];
    const float* enc_muW1 = (const float*)d_in[9];
    const float* enc_muB1 = (const float*)d_in[10];
    const float* enc_muW2 = (const float*)d_in[11];
    const float* enc_muB2 = (const float*)d_in[12];
    const float* enc_vaW1 = (const float*)d_in[13];
    const float* enc_vaB1 = (const float*)d_in[14];
    const float* enc_vaW2 = (const float*)d_in[15];
    const float* enc_vaB2 = (const float*)d_in[16];
    const float* dec_cw1  = (const float*)d_in[17];
    const float* dec_cb1  = (const float*)d_in[18];
    const float* dec_cw2  = (const float*)d_in[19];
    const float* dec_cb2  = (const float*)d_in[20];
    const float* dec_cw3  = (const float*)d_in[21];
    const float* dec_cb3  = (const float*)d_in[22];

    char* base = (char*)d_ws;
    size_t off = 0;
    auto take = [&](size_t bytes) -> char* {
        char* p = base + off;
        off += (bytes + 255) & ~(size_t)255;
        return p;
    };
    float*  A    = (float*) take((size_t)P_ * 64 * 4);   // pre-BN conv acts / split-K partials
    __bf16* B    = (__bf16*)take((size_t)P_ * 64 * 2);   // post-BN acts / flattened f (bf16)
    __bf16* W1MU = (__bf16*)take((size_t)H_ * F_ * 2);
    __bf16* W1VA = (__bf16*)take((size_t)H_ * F_ * 2);
    __bf16* W2MU = (__bf16*)take((size_t)2 * L_ * H_ * 2);
    __bf16* W2VA = (__bf16*)take((size_t)2 * L_ * H_ * 2);
    __bf16* WC2E = (__bf16*)take((size_t)2 * 9 * 64 * 32 * 2);
    __bf16* WC3E = (__bf16*)take((size_t)2 * 9 * 32 * 64 * 2);
    __bf16* WC2D = (__bf16*)take((size_t)2 * 9 * 64 * 32 * 2);
    float*  MU0  = (float*) take((size_t)N_ * L_ * 4);
    float*  MU1  = (float*) take((size_t)N_ * L_ * 4);
    float*  LV0  = (float*) take((size_t)N_ * L_ * 4);
    float*  LV1  = (float*) take((size_t)N_ * L_ * 4);
    __bf16* HID  = (__bf16*)take((size_t)N_ * H_ * 2);
    float*  ZC   = (float*) take((size_t)N_ * L_ * 4);
    float*  ZS   = (float*) take((size_t)N_ * L_ * 4);
    float*  OUTC = (float*) take((size_t)N_ * L_ * 4);
    float*  OUTS = (float*) take((size_t)N_ * L_ * 4);
    float*  Y3D  = (float*) take((size_t)P_ * 4);
    float*  GMU  = (float*) take((size_t)L_ * 4);
    float*  GLV  = (float*) take((size_t)L_ * 4);
    float*  STATS= (float*) take(64 * 2 * 4);
    float*  SCAL = (float*) take(256);

    float* PART = A;               // split-K partials reuse the big f32 buffer
    const int CW2 = 9 * 64 * 32;
    const int CW3 = 9 * 32 * 64;

    // ---- weight conversion ----------------------------------------------
    k_cvt<<<(2 * L_ * H_) / 256, 256, 0, stream>>>(enc_muW2, W2MU, 2 * L_ * H_);
    k_cvt<<<(2 * L_ * H_) / 256, 256, 0, stream>>>(enc_vaW2, W2VA, 2 * L_ * H_);
    for (int i = 0; i < 2; ++i) {
        k_cvt_convw<<<(CW2 + 255) / 256, 256, 0, stream>>>(enc_cw2 + (size_t)i * CW2, WC2E + (size_t)i * CW2, 64, 32);
        k_cvt_convw<<<(CW3 + 255) / 256, 256, 0, stream>>>(enc_cw3 + (size_t)i * CW3, WC3E + (size_t)i * CW3, 32, 64);
        k_cvt_convw<<<(CW2 + 255) / 256, 256, 0, stream>>>(dec_cw2 + (size_t)i * CW2, WC2D + (size_t)i * CW2, 64, 32);
    }
    k_zero<<<1, 64, 0, stream>>>(SCAL, 2);

    auto bn_stats = [&](const float* y, int C) {
        k_zero<<<1, 256, 0, stream>>>(STATS, 128);
        k_stats<<<2048, 256, 0, stream>>>(y, C, STATS);
    };

    // ---- encoders (i = 0 class, i = 1 style) ----------------------------
    float* MU[2] = {MU0, MU1};
    float* LV[2] = {LV0, LV1};
    for (int i = 0; i < 2; ++i) {
        k_conv1<<<P_ / 256, 256, 0, stream>>>(x, enc_cw1 + (size_t)i * 32 * 9, enc_cb1 + i * 32, A);
        bn_stats(A, 32);
        k_bn_lrelu<true><<<(P_ * 32) / 256, 256, 0, stream>>>(A, STATS, 31, B, (size_t)P_ * 32);
        k_conv2_wmma<<<P_ / 32, 32, 0, stream>>>(B, WC2E + (size_t)i * CW2, enc_cb2 + i * 64, A);
        bn_stats(A, 64);
        k_bn_lrelu<true><<<(P_ * 64) / 256, 256, 0, stream>>>(A, STATS, 63, B, (size_t)P_ * 64);
        k_conv3_wmma<<<P_ / 32, 32, 0, stream>>>(B, WC3E + (size_t)i * CW3, enc_cb3 + i * 32, A);
        bn_stats(A, 32);
        k_bn_lrelu<true><<<(P_ * 32) / 256, 256, 0, stream>>>(A, STATS, 31, B, (size_t)P_ * 32);
        // MLP heads: split-K WMMA GEMMs. B == f [512, 32768] bf16 (NHWC flatten).
        k_cvt_w1<<<H_ * 32, 256, 0, stream>>>(enc_muW1 + (size_t)i * H_ * F_, W1MU);
        k_cvt_w1<<<H_ * 32, 256, 0, stream>>>(enc_vaW1 + (size_t)i * H_ * F_, W1VA);
        // mu head
        k_gemm_splitk<<<dim3(N_ / 32, H_ / 64, 8), 32, 0, stream>>>(B, W1MU, PART, N_, H_, F_ / 8);
        k_gemm_fin<true><<<(N_ * H_) / 256, 256, 0, stream>>>(PART, enc_muB1 + i * H_, HID, N_, H_, H_ - 1, 8);
        k_gemm_splitk<<<dim3(N_ / 32, L_ / 64, 4), 32, 0, stream>>>(HID, W2MU + (size_t)i * L_ * H_, PART, N_, L_, H_ / 4);
        k_gemm_fin<false><<<(N_ * L_) / 256, 256, 0, stream>>>(PART, enc_muB2 + i * L_, MU[i], N_, L_, L_ - 1, 4);
        // logvar head
        k_gemm_splitk<<<dim3(N_ / 32, H_ / 64, 8), 32, 0, stream>>>(B, W1VA, PART, N_, H_, F_ / 8);
        k_gemm_fin<true><<<(N_ * H_) / 256, 256, 0, stream>>>(PART, enc_vaB1 + i * H_, HID, N_, H_, H_ - 1, 8);
        k_gemm_splitk<<<dim3(N_ / 32, L_ / 64, 4), 32, 0, stream>>>(HID, W2VA + (size_t)i * L_ * H_, PART, N_, L_, H_ / 4);
        k_gemm_fin<false><<<(N_ * L_) / 256, 256, 0, stream>>>(PART, enc_vaB2 + i * L_, LV[i], N_, L_, L_ - 1, 4);
    }

    // ---- PoE group posterior + reparameterize ---------------------------
    k_poe<<<L_, 256, 0, stream>>>(MU1, LV1, GMU, GLV);
    k_reparam<<<(N_ * L_) / 256, 256, 0, stream>>>(eps_c, eps_s, MU0, LV0, GMU, GLV, ZC, ZS);

    // ---- decoders -------------------------------------------------------
    const float* Z[2]   = {ZC, ZS};
    float*       OUT[2] = {OUTC, OUTS};
    for (int i = 0; i < 2; ++i) {
        k_conv1<<<P_ / 256, 256, 0, stream>>>(Z[i], dec_cw1 + (size_t)i * 32 * 9, dec_cb1 + i * 32, A);
        bn_stats(A, 32);
        k_bn_lrelu<true><<<(P_ * 32) / 256, 256, 0, stream>>>(A, STATS, 31, B, (size_t)P_ * 32);
        k_conv2_wmma<<<P_ / 32, 32, 0, stream>>>(B, WC2D + (size_t)i * CW2, dec_cb2 + i * 64, A);
        bn_stats(A, 64);
        k_bn_lrelu<true><<<(P_ * 64) / 256, 256, 0, stream>>>(A, STATS, 63, B, (size_t)P_ * 64);
        k_conv3_dec<<<P_ / 256, 256, 0, stream>>>(B, dec_cw3 + (size_t)i * 64 * 9, dec_cb3 + i, Y3D);
        bn_stats(Y3D, 1);
        k_bn_lrelu<false><<<P_ / 256, 256, 0, stream>>>(Y3D, STATS, 0, OUT[i], (size_t)P_);
    }

    // ---- losses + final elementwise ------------------------------------
    k_kl1<<<256, 256, 0, stream>>>(MU0, LV0, SCAL);
    k_kl2<<<1, 256, 0, stream>>>(GMU, GLV, SCAL);
    k_final<<<P_ / 256, 256, 0, stream>>>(x, OUTC, OUTS, SCAL, (float*)d_out);
}
